// AdaptiveTabularMoELayer_28260884807821
// MI455X (gfx1250) — compile-verified, matrix operand order checked by reading, not run on hardware
//
#include <hip/hip_runtime.h>
#include <hip/hip_bf16.h>

typedef __attribute__((ext_vector_type(16))) _Float16 v16h;
typedef __attribute__((ext_vector_type(8)))  float    v8f;

#define N_TOK   (128 * 2048)          // B*S tokens
#define DK      128                   // feature dim
#define HH      64                    // hidden 1
#define H2      32                    // hidden 2
#define EE      8                     // experts

// d_out layout (floats): routing_weights [N_TOK*8] | predicted_types [N_TOK*3] | loss | entropy
#define OFF_RW   0
#define OFF_PT   (N_TOK * 8)
#define OFF_LOSS (N_TOK * 8 + N_TOK * 3)
#define OFF_ENT  (OFF_LOSS + 1)

// ws layout (bytes):
//   0      : float acc[16]            (8 expert sums, 1 entropy sum, pad)
//   64     : _Float16 w1cm[64*128]    col-major gw1 (16384 B)
//   16448  : _Float16 w2cm[32*64]     col-major gw2 (4096 B)
//   20544  : _Float16 w3cm[16*32]     col-major gw3 padded to 16 cols (1024 B)
//   21568  : float tbias[3*8]         precomputed type_emb@tpw + tpb
#define WS_W1  64
#define WS_W2  (WS_W1 + 64 * 128 * 2)
#define WS_W3  (WS_W2 + 32 * 64 * 2)
#define WS_TB  (WS_W3 + 16 * 32 * 2)

// ---------------------------------------------------------------------------
// Preprocess: fp32 weights -> f16 column-major fragments, 3x8 type-bias table
// ---------------------------------------------------------------------------
__global__ void moe_prep_kernel(const float* __restrict__ gw1,
                                const float* __restrict__ gw2,
                                const float* __restrict__ gw3,
                                const float* __restrict__ type_emb,
                                const float* __restrict__ tpw,
                                const float* __restrict__ tpb,
                                _Float16* __restrict__ w1cm,
                                _Float16* __restrict__ w2cm,
                                _Float16* __restrict__ w3cm,
                                float* __restrict__ tbias) {
  const int t = threadIdx.x;
  for (int i = t; i < HH * DK; i += 256) {         // w1cm[n*128+k] = gw1[k][n]
    int n = i / DK, k = i % DK;
    w1cm[i] = (_Float16)gw1[k * HH + n];
  }
  for (int i = t; i < H2 * HH; i += 256) {         // w2cm[n*64+k] = gw2[k][n]
    int n = i / HH, k = i % HH;
    w2cm[i] = (_Float16)gw2[k * H2 + n];
  }
  for (int i = t; i < 16 * H2; i += 256) {         // w3cm padded to 16 cols
    int n = i / H2, k = i % H2;
    w3cm[i] = (n < EE) ? (_Float16)gw3[k * EE + n] : (_Float16)0.0f;
  }
  if (t < 24) {                                    // tbias[ty][e] = te@tpw + tpb
    int ty = t / EE, e = t % EE;
    float s = tpb[e];
    for (int j = 0; j < 16; ++j) s += type_emb[ty * 16 + j] * tpw[j * EE + e];
    tbias[t] = s;
  }
}

// ---------------------------------------------------------------------------
// Main router kernel: 8 waves / WG, each wave does 8 tiles of 16 tokens.
// 21 v_wmma_f32_16x16x32_f16 per tile (16 for layer1, 4 for layer2, 1 for layer3)
// ---------------------------------------------------------------------------
__global__ __launch_bounds__(256) void moe_router_kernel(
    const float* __restrict__ x, const int* __restrict__ ftypes,
    const float* __restrict__ gb1, const float* __restrict__ gb2,
    const float* __restrict__ gb3,
    const _Float16* __restrict__ w1cm, const _Float16* __restrict__ w2cm,
    const _Float16* __restrict__ w3cm, const float* __restrict__ tbias,
    float* __restrict__ out_rw, float* __restrict__ out_pt,
    float* __restrict__ acc) {
  __shared__ _Float16 lds_h[8][16 * HH] __attribute__((aligned(32)));
  __shared__ _Float16 lds_h2[8][16 * H2] __attribute__((aligned(32)));
  __shared__ float    lds_lg[8][16 * EE];

  const int wv   = threadIdx.x >> 5;
  const int lane = threadIdx.x & 31;
  const int lo   = lane & 15;
  const int hi   = lane >> 4;   // 0: K/M 0..15 half, 1: K 16..31 / M 8..15 half

  // ---- B fragments in VGPRs (weights stay L2/L0-resident, loaded once) ----
  v16h B1[4][4];
#pragma unroll
  for (int nt = 0; nt < 4; ++nt)
#pragma unroll
    for (int ch = 0; ch < 4; ++ch)
      B1[nt][ch] = *(const v16h*)(w1cm + (nt * 16 + lo) * DK + ch * 32 + hi * 16);
  v16h B2[2][2];
#pragma unroll
  for (int nt = 0; nt < 2; ++nt)
#pragma unroll
    for (int ch = 0; ch < 2; ++ch)
      B2[nt][ch] = *(const v16h*)(w2cm + (nt * 16 + lo) * HH + ch * 32 + hi * 16);
  v16h B3 = *(const v16h*)(w3cm + lo * H2 + hi * 16);

  float bias1[4];
#pragma unroll
  for (int nt = 0; nt < 4; ++nt) bias1[nt] = gb1[nt * 16 + lo];
  float bias2[2];
#pragma unroll
  for (int nt = 0; nt < 2; ++nt) bias2[nt] = gb2[nt * 16 + lo];
  const float bias3 = (lo < EE) ? gb3[lo] : 0.0f;

  float aex[EE] = {0.f, 0.f, 0.f, 0.f, 0.f, 0.f, 0.f, 0.f};
  float aent = 0.0f;

  const int wg_base = blockIdx.x * 1024;   // 8 waves * 8 tiles * 16 tokens
#pragma unroll 1
  for (int t = 0; t < 8; ++t) {
    const int tile = wg_base + (t * 8 + wv) * 16;
    const float* xrow = x + (size_t)(tile + lo) * DK + hi * 16;
    if (t < 7) __builtin_prefetch(xrow + 128 * DK, 0, 1);  // next tile of this wave

    // ---- A fragments: 16 contiguous fp32 per lane -> f16 (contiguous k = 16*hi + s, same order as B1) ----
    v16h A[4];
#pragma unroll
    for (int ch = 0; ch < 4; ++ch) {
      const float4* p = (const float4*)(xrow + ch * 32);
      float4 f0 = p[0], f1 = p[1], f2 = p[2], f3 = p[3];
      v16h a;
      a[0]  = (_Float16)f0.x; a[1]  = (_Float16)f0.y; a[2]  = (_Float16)f0.z; a[3]  = (_Float16)f0.w;
      a[4]  = (_Float16)f1.x; a[5]  = (_Float16)f1.y; a[6]  = (_Float16)f1.z; a[7]  = (_Float16)f1.w;
      a[8]  = (_Float16)f2.x; a[9]  = (_Float16)f2.y; a[10] = (_Float16)f2.z; a[11] = (_Float16)f2.w;
      a[12] = (_Float16)f3.x; a[13] = (_Float16)f3.y; a[14] = (_Float16)f3.z; a[15] = (_Float16)f3.w;
      A[ch] = a;
    }

    // ---- Layer 1: [16x128] @ [128x64], relu, stage to LDS as f16 ----
    _Float16* hbuf = lds_h[wv];
#pragma unroll
    for (int nt = 0; nt < 4; ++nt) {
      v8f c = {};
#pragma unroll
      for (int ch = 0; ch < 4; ++ch)
        c = __builtin_amdgcn_wmma_f32_16x16x32_f16(false, A[ch], false,
                                                   B1[nt][ch], (short)0, c,
                                                   false, false);
      const int n = nt * 16 + lo;
#pragma unroll
      for (int r = 0; r < 8; ++r) {
        float v = c[r] + bias1[nt];
        v = v > 0.0f ? v : 0.0f;
        hbuf[(r + hi * 8) * HH + n] = (_Float16)v;   // row-major h[16][64]
      }
    }

    // ---- Layer 2: [16x64] @ [64x32], relu ----
    v16h A2[2];
#pragma unroll
    for (int ch = 0; ch < 2; ++ch)
      A2[ch] = *(const v16h*)(hbuf + lo * HH + ch * 32 + hi * 16);
    _Float16* h2buf = lds_h2[wv];
#pragma unroll
    for (int nt = 0; nt < 2; ++nt) {
      v8f c = {};
      c = __builtin_amdgcn_wmma_f32_16x16x32_f16(false, A2[0], false, B2[nt][0],
                                                 (short)0, c, false, false);
      c = __builtin_amdgcn_wmma_f32_16x16x32_f16(false, A2[1], false, B2[nt][1],
                                                 (short)0, c, false, false);
      const int n = nt * 16 + lo;
#pragma unroll
      for (int r = 0; r < 8; ++r) {
        float v = c[r] + bias2[nt];
        v = v > 0.0f ? v : 0.0f;
        h2buf[(r + hi * 8) * H2 + n] = (_Float16)v;
      }
    }

    // ---- Layer 3: [16x32] @ [32x8 padded to 16] ----
    v16h A3 = *(const v16h*)(h2buf + lo * H2 + hi * 16);
    v8f c3 = {};
    c3 = __builtin_amdgcn_wmma_f32_16x16x32_f16(false, A3, false, B3, (short)0,
                                                c3, false, false);
    float* lg = lds_lg[wv];
    if (lo < EE) {
#pragma unroll
      for (int r = 0; r < 8; ++r)
        lg[(r + hi * 8) * EE + lo] = c3[r] + bias3;
    }

    // ---- Per-token tail: lanes 0..15 each own one token of the tile ----
    if (lane < 16) {
      const int tok = tile + lane;
      const int ty  = ftypes[tok];
      float l8[EE];
#pragma unroll
      for (int e = 0; e < EE; ++e)
        l8[e] = lg[lane * EE + e] + tbias[ty * EE + e];
      // softmax #1 (primary)
      float mx = l8[0];
#pragma unroll
      for (int e = 1; e < EE; ++e) mx = fmaxf(mx, l8[e]);
      float s = 0.0f;
#pragma unroll
      for (int e = 0; e < EE; ++e) { l8[e] = __expf(l8[e] - mx); s += l8[e]; }
      const float inv = 1.0f / s;
      // + TYPE_BIAS (0.5 on matching-type experts) ; experts: 0-2 num, 3-5 cat, 6-7 int
#pragma unroll
      for (int e = 0; e < EE; ++e) {
        const int et = (e < 3) ? 0 : (e < 6) ? 1 : 2;
        l8[e] = l8[e] * inv + ((et == ty) ? 0.5f : 0.0f);
      }
      // softmax #2 (final routing weights)
      float mx2 = l8[0];
#pragma unroll
      for (int e = 1; e < EE; ++e) mx2 = fmaxf(mx2, l8[e]);
      float s2 = 0.0f;
#pragma unroll
      for (int e = 0; e < EE; ++e) { l8[e] = __expf(l8[e] - mx2); s2 += l8[e]; }
      const float inv2 = 1.0f / s2;
#pragma unroll
      for (int e = 0; e < EE; ++e) l8[e] *= inv2;
      // outputs
      float4* o = (float4*)(out_rw + (size_t)tok * EE);
      o[0] = make_float4(l8[0], l8[1], l8[2], l8[3]);
      o[1] = make_float4(l8[4], l8[5], l8[6], l8[7]);
      float* pt = out_pt + (size_t)tok * 3;
      pt[0] = (ty == 0) ? 1.0f : 0.0f;
      pt[1] = (ty == 1) ? 1.0f : 0.0f;
      pt[2] = (ty == 2) ? 1.0f : 0.0f;
      // metric partials
#pragma unroll
      for (int e = 0; e < EE; ++e) {
        aex[e] += l8[e];
        aent   -= l8[e] * __logf(l8[e] + 1e-8f);
      }
    }
  }

  // ---- reduce partials across lanes 0..15 (16..31 hold zeros), one atomic set per wave
#pragma unroll
  for (int off = 1; off < 16; off <<= 1) {
#pragma unroll
    for (int e = 0; e < EE; ++e) aex[e] += __shfl_xor(aex[e], off, 32);
    aent += __shfl_xor(aent, off, 32);
  }
  if (lane == 0) {
#pragma unroll
    for (int e = 0; e < EE; ++e) atomicAdd(&acc[e], aex[e]);
    atomicAdd(&acc[8], aent);
  }
}

// ---------------------------------------------------------------------------
// Finalize: load-balance loss (unbiased var * E^2 * 0.01) and mean entropy
// ---------------------------------------------------------------------------
__global__ void moe_finalize_kernel(const float* __restrict__ acc,
                                    float* __restrict__ out) {
  if (threadIdx.x == 0 && blockIdx.x == 0) {
    const float invN = 1.0f / (float)N_TOK;
    float f[EE], mean = 0.0f;
    for (int e = 0; e < EE; ++e) { f[e] = acc[e] * invN; mean += f[e]; }
    mean *= (1.0f / EE);
    float var = 0.0f;
    for (int e = 0; e < EE; ++e) { float d = f[e] - mean; var += d * d; }
    var *= (1.0f / (EE - 1));                 // torch unbiased var
    out[OFF_LOSS] = 0.01f * var * (float)(EE * EE);
    out[OFF_ENT]  = acc[8] * invN;
  }
}

extern "C" void kernel_launch(void* const* d_in, const int* in_sizes, int n_in,
                              void* d_out, int out_size, void* d_ws, size_t ws_size,
                              hipStream_t stream) {
  const float* x        = (const float*)d_in[0];
  const int*   ftypes   = (const int*)d_in[1];
  const float* gw1      = (const float*)d_in[2];
  const float* gb1      = (const float*)d_in[3];
  const float* gw2      = (const float*)d_in[4];
  const float* gb2      = (const float*)d_in[5];
  const float* gw3      = (const float*)d_in[6];
  const float* gb3      = (const float*)d_in[7];
  const float* type_emb = (const float*)d_in[8];
  const float* tpw      = (const float*)d_in[9];
  const float* tpb      = (const float*)d_in[10];
  // d_in[11..14] (secondary router) are dead: FEATURE_TYPE_WEIGHT == 1.0

  char* ws = (char*)d_ws;
  float*     acc   = (float*)ws;
  _Float16*  w1cm  = (_Float16*)(ws + WS_W1);
  _Float16*  w2cm  = (_Float16*)(ws + WS_W2);
  _Float16*  w3cm  = (_Float16*)(ws + WS_W3);
  float*     tbias = (float*)(ws + WS_TB);

  float* out    = (float*)d_out;
  float* out_rw = out + OFF_RW;
  float* out_pt = out + OFF_PT;

  hipMemsetAsync(acc, 0, 16 * sizeof(float), stream);
  moe_prep_kernel<<<1, 256, 0, stream>>>(gw1, gw2, gw3, type_emb, tpw, tpb,
                                         w1cm, w2cm, w3cm, tbias);
  moe_router_kernel<<<256, 256, 0, stream>>>(x, ftypes, gb1, gb2, gb3,
                                             w1cm, w2cm, w3cm, tbias,
                                             out_rw, out_pt, acc);
  moe_finalize_kernel<<<1, 64, 0, stream>>>(acc, out);
}
